// ResNet_15633680958099
// MI455X (gfx1250) — matrix operand order repaired and checked by reading, not compiled, not run on hardware
//
#include <hip/hip_runtime.h>
#include <hip/hip_bf16.h>
#include <cstdint>
#include <cstddef>

// ---------------------------------------------------------------------------
// CDNA5 / gfx1250 3D-ResNet forward.
// Convs = implicit GEMM on v_wmma_f32_16x16x32_bf16 with channels-last
// (NDHWC) activations and [Cout][k^3][Cin] weights.  Each wave computes a
// 32x32 output macro-tile (2x2 WMMA register blocking): per 32-K chunk,
// 8 aligned b128 loads feed 4 WMMAs (2 loads per WMMA).
// ---------------------------------------------------------------------------

typedef __attribute__((ext_vector_type(16))) __bf16 v16bf;
typedef __attribute__((ext_vector_type(8)))  __bf16 v8bf;
typedef __attribute__((ext_vector_type(8)))  float  v8f;

#define GRID3 216000   // 60^3
#define NATOM 512

__device__ __forceinline__ __bf16 f2bf(float f) {
  unsigned u = __builtin_bit_cast(unsigned, f);
  unsigned r = (u + 0x7FFFu + ((u >> 16) & 1u)) >> 16;   // RNE
  return __builtin_bit_cast(__bf16, (unsigned short)r);
}

__device__ __forceinline__ v16bf cat16(v8bf lo, v8bf hi) {
  return __builtin_shufflevector(lo, hi, 0, 1, 2, 3, 4, 5, 6, 7,
                                 8, 9, 10, 11, 12, 13, 14, 15);
}

// ---------------------------------------------------------------------------
// Gaussian blurring
// ---------------------------------------------------------------------------
__global__ void colsum_kernel(const float* __restrict__ xin, float* __restrict__ colsum) {
  const int a = blockIdx.x;
  const float ax = xin[a * 5 + 2], ay = xin[a * 5 + 3], az = xin[a * 5 + 4];
  const float factor = 15.432098765432098f;   // 1/(2*(0.6*0.3)^2)
  float s = 0.f;
  for (int g = threadIdx.x; g < GRID3; g += blockDim.x) {
    int gx = g / 3600, gy = (g / 60) % 60, gz = g % 60;
    float dx = (-8.85f + 0.3f * gx) - ax;
    float dy = (-8.85f + 0.3f * gy) - ay;
    float dz = (-8.85f + 0.3f * gz) - az;
    s += __expf(-(dx * dx + dy * dy + dz * dz) * factor);
  }
  __shared__ float red[256];
  red[threadIdx.x] = s;
  __syncthreads();
  for (int st = 128; st > 0; st >>= 1) {
    if ((int)threadIdx.x < st) red[threadIdx.x] += red[threadIdx.x + st];
    __syncthreads();
  }
  if (threadIdx.x == 0) colsum[a] = fmaxf(red[0], 1e-6f);
}

// One thread per grid point; 16 segment accumulators (batch*atom_type).
// Emits conv1 input in NDHWC, channel-padded 4 -> 32: [B=4][60^3][32] bf16.
__global__ void fields_kernel(const float* __restrict__ xin,
                              const float* __restrict__ colsum,
                              __bf16* __restrict__ fieldsBF) {
  __shared__ float sax[NATOM], say[NATOM], saz[NATOM], sinv[NATOM];
  __shared__ int   sseg[NATOM];
  for (int a = threadIdx.x; a < NATOM; a += blockDim.x) {
    sax[a] = xin[a * 5 + 2];
    say[a] = xin[a * 5 + 3];
    saz[a] = xin[a * 5 + 4];
    sseg[a] = (int)xin[a * 5 + 0] * 4 + (int)xin[a * 5 + 1];
    sinv[a] = 1.0f / colsum[a];
  }
  __syncthreads();
  const int g = blockIdx.x * blockDim.x + threadIdx.x;
  if (g >= GRID3) return;
  const int gx = g / 3600, gy = (g / 60) % 60, gz = g % 60;
  const float px = -8.85f + 0.3f * gx;
  const float py = -8.85f + 0.3f * gy;
  const float pz = -8.85f + 0.3f * gz;
  const float factor = 15.432098765432098f;
  float acc[16];
#pragma unroll
  for (int s = 0; s < 16; ++s) acc[s] = 0.f;
  for (int a = 0; a < NATOM; ++a) {
    float dx = px - sax[a], dy = py - say[a], dz = pz - saz[a];
    float d2 = dx * dx + dy * dy + dz * dz;
    acc[sseg[a]] += __expf(-d2 * factor) * sinv[a];
  }
  const __bf16 bz = __builtin_bit_cast(__bf16, (unsigned short)0);
  for (int bb = 0; bb < 4; ++bb) {
    __bf16* p = fieldsBF + ((size_t)bb * GRID3 + g) * 32;
#pragma unroll
    for (int c = 0; c < 4; ++c) p[c] = f2bf(acc[bb * 4 + c]);
#pragma unroll
    for (int c = 4; c < 32; ++c) p[c] = bz;
  }
}

// ---------------------------------------------------------------------------
// Weight staging: OIDHW f32 -> [Cout][k^3][CinP] bf16 (zero-pad ci >= Cin).
// ---------------------------------------------------------------------------
__global__ void stage_weights_kernel(const float* __restrict__ src, __bf16* __restrict__ dst,
                                     int Cout, int Cin, int CinP, int k3) {
  const int total = Cout * k3 * CinP;
  const int i = blockIdx.x * blockDim.x + threadIdx.x;
  if (i >= total) return;
  const int ci = i % CinP;
  const int t  = (i / CinP) % k3;
  const int m  = i / (CinP * k3);
  float v = (ci < Cin) ? src[((size_t)m * Cin + ci) * k3 + t] : 0.f;
  dst[i] = f2bf(v);
}

// ---------------------------------------------------------------------------
// Implicit-GEMM conv3d on WMMA (bf16 in, f32 acc), 2x2 register blocking.
//   Activations: NDHWC [B][D][H][W][Cin], Cin multiple of 32.
//   Weights:     [Cout][k^3][Cin] (K order = tap-major, channel-minor).
//   Output:      NDHWC [B][Dout^3][Cout] f32.  Cout multiple of 32.
// One wave per 32x32 output macro-tile.  Per 32-K chunk: 2 A-tile loads +
// 2 B-tile loads (4x b128 each pair) feed 4 WMMAs into 4 accumulators.
// CDNA5 16-bit operand layouts (ISA 7.12.2):
//   A: lane ci-offsets {0..7}u{16..23} + 8*(lane>=16)
//   B: lane ci-offsets {0..7}u{16..23} + 8*(lane>=16),  N = lane&15
//   C/D: M = v + 8*(lane>=16), N = lane&15
// ---------------------------------------------------------------------------
__global__ void conv3d_wmma_kernel(const __bf16* __restrict__ X,
                                   const __bf16* __restrict__ W,
                                   float* __restrict__ Y,
                                   int Bn, int Cin, int Din, int Cout, int Dout,
                                   int ksz, int stride, int pad) {
  const int lane    = threadIdx.x & 31;
  const int mtiles2 = Cout >> 5;                      // 32-wide M macro-tiles
  const int Dout2   = Dout * Dout;
  const int Dout3   = Dout2 * Dout;
  const int Ntot    = Bn * Dout3;
  const int ntiles2 = (Ntot + 31) >> 5;               // 32-wide N macro-tiles
  const int tile    = blockIdx.x * (blockDim.x >> 5) + (threadIdx.x >> 5);
  if (tile >= mtiles2 * ntiles2) return;              // wave-uniform
  const int mtile2 = tile % mtiles2;
  const int ntile2 = tile / mtiles2;
  const int k3     = ksz * ksz * ksz;

  const int l15   = lane & 15;
  const int kAoff = (lane >= 16) ? 8 : 0;
  const int kBoff = (lane >= 16) ? 8 : 0;

  // ---- N side: two 16-column subtiles ----
  int  ncol[2];
  bool nval[2];
  int  idb[2], ihb[2], iwb[2];
  const __bf16* Xb[2];
#pragma unroll
  for (int nt = 0; nt < 2; ++nt) {
    ncol[nt] = ntile2 * 32 + nt * 16 + l15;
    nval[nt] = ncol[nt] < Ntot;
    int b = 0, id0 = 0, ih0 = 0, iw0 = 0;
    if (nval[nt]) {
      b = ncol[nt] / Dout3;
      int r  = ncol[nt] - b * Dout3;
      int od = r / Dout2; r -= od * Dout2;
      int oh = r / Dout;
      int ow = r - oh * Dout;
      id0 = od * stride - pad;
      ih0 = oh * stride - pad;
      iw0 = ow * stride - pad;
    }
    idb[nt] = id0; ihb[nt] = ih0; iwb[nt] = iw0;
    Xb[nt]  = X + ((size_t)b * Din * Din * Din) * Cin + kBoff;
  }

  // ---- M side: two 16-row subtiles (always in range: Cout mult of 32) ----
  const __bf16* Wrow[2];
#pragma unroll
  for (int mt = 0; mt < 2; ++mt) {
    int m = mtile2 * 32 + mt * 16 + l15;
    Wrow[mt] = W + ((size_t)m * k3) * Cin + kAoff;
  }

  v16bf zerob;
#pragma unroll
  for (int i = 0; i < 16; ++i) zerob[i] = __builtin_bit_cast(__bf16, (unsigned short)0);

  v8f acc[2][2];
#pragma unroll
  for (int mt = 0; mt < 2; ++mt)
#pragma unroll
    for (int nt = 0; nt < 2; ++nt)
#pragma unroll
      for (int v = 0; v < 8; ++v) acc[mt][nt][v] = 0.f;

  int kd = 0, kh = 0, kw = 0;
  for (int t = 0; t < k3; ++t) {
    bool valid[2];
    const __bf16* pB[2];
#pragma unroll
    for (int nt = 0; nt < 2; ++nt) {
      const int id = idb[nt] + kd, ih = ihb[nt] + kh, iw = iwb[nt] + kw;
      valid[nt] = nval[nt] && (unsigned)id < (unsigned)Din &&
                  (unsigned)ih < (unsigned)Din && (unsigned)iw < (unsigned)Din;
      pB[nt] = Xb[nt] + ((size_t)(id * Din + ih) * Din + iw) * (size_t)Cin;
    }
    const __bf16* pA0 = Wrow[0] + (size_t)t * Cin;
    const __bf16* pA1 = Wrow[1] + (size_t)t * Cin;
    if (t + 1 < k3) __builtin_prefetch((const void*)(pA0 + Cin), 0, 1);

    for (int c0 = 0; c0 < Cin; c0 += 32) {
      v16bf amat[2], bmat[2];
      amat[0] = cat16(*(const v8bf*)(pA0 + c0), *(const v8bf*)(pA0 + c0 + 16));
      amat[1] = cat16(*(const v8bf*)(pA1 + c0), *(const v8bf*)(pA1 + c0 + 16));
#pragma unroll
      for (int nt = 0; nt < 2; ++nt) {
        bmat[nt] = zerob;
        if (valid[nt])
          bmat[nt] = cat16(*(const v8bf*)(pB[nt] + c0), *(const v8bf*)(pB[nt] + c0 + 16));
      }
      // 4 independent accumulators: no serial WMMA dependency inside a chunk.
      acc[0][0] = __builtin_amdgcn_wmma_f32_16x16x32_bf16(false, amat[0], false, bmat[0],
                                                          (short)0, acc[0][0], false, false);
      acc[0][1] = __builtin_amdgcn_wmma_f32_16x16x32_bf16(false, amat[0], false, bmat[1],
                                                          (short)0, acc[0][1], false, false);
      acc[1][0] = __builtin_amdgcn_wmma_f32_16x16x32_bf16(false, amat[1], false, bmat[0],
                                                          (short)0, acc[1][0], false, false);
      acc[1][1] = __builtin_amdgcn_wmma_f32_16x16x32_bf16(false, amat[1], false, bmat[1],
                                                          (short)0, acc[1][1], false, false);
    }
    if (++kw == ksz) { kw = 0; if (++kh == ksz) { kh = 0; ++kd; } }
  }

  // ---- store: per lane 8 contiguous f32 per (mt, nt) subtile ----
  const int mlane = (lane >= 16) ? 8 : 0;
#pragma unroll
  for (int nt = 0; nt < 2; ++nt) {
    if (nval[nt]) {
#pragma unroll
      for (int mt = 0; mt < 2; ++mt) {
        float* Yp = Y + (size_t)ncol[nt] * Cout + mtile2 * 32 + mt * 16 + mlane;
        *(v8f*)Yp = acc[mt][nt];   // 32B-aligned
      }
    }
  }
}

// ---------------------------------------------------------------------------
// BatchNorm statistics (NDHWC): per-channel mean and rsqrt(var + eps).
// ---------------------------------------------------------------------------
__global__ void bn_stats_kernel(const float* __restrict__ X, int Bn, int C, int S,
                                float* __restrict__ stats) {
  const int c = blockIdx.x;
  const int n = Bn * S;
  float s1 = 0.f, s2 = 0.f;
  for (int i = threadIdx.x; i < n; i += blockDim.x) {
    float v = X[(size_t)i * C + c];
    s1 += v;
    s2 += v * v;
  }
  __shared__ float r1[256], r2[256];
  r1[threadIdx.x] = s1;
  r2[threadIdx.x] = s2;
  __syncthreads();
  for (int st = 128; st > 0; st >>= 1) {
    if ((int)threadIdx.x < st) {
      r1[threadIdx.x] += r1[threadIdx.x + st];
      r2[threadIdx.x] += r2[threadIdx.x + st];
    }
    __syncthreads();
  }
  if (threadIdx.x == 0) {
    float mm  = r1[0] / (float)n;
    float var = r2[0] / (float)n - mm * mm;
    stats[2 * c]     = mm;
    stats[2 * c + 1] = rsqrtf(var + 1e-5f);
  }
}

// BN + affine + optional residual + activation (NDHWC).  act:0 none,1 relu,2 leaky0.1
__global__ void bn_act_kernel(const float* __restrict__ X,
                              const float* __restrict__ gamma,
                              const float* __restrict__ beta,
                              const float* __restrict__ stats,
                              const float* __restrict__ resid,
                              float* __restrict__ Yf, __bf16* __restrict__ Ybf,
                              int C, int total, int act) {
  const int idx = blockIdx.x * blockDim.x + threadIdx.x;
  if (idx >= total) return;
  const int c = idx % C;
  float v = (X[idx] - stats[2 * c]) * stats[2 * c + 1] * gamma[c] + beta[c];
  if (resid) v += resid[idx];
  if (act == 1) v = fmaxf(v, 0.f);
  else if (act == 2) v = (v > 0.f) ? v : 0.1f * v;
  Yf[idx] = v;
  if (Ybf) Ybf[idx] = f2bf(v);
}

// maxpool 3x3x3 stride 2 pad 1 (NDHWC)
__global__ void maxpool3_kernel(const float* __restrict__ X, float* __restrict__ Yf,
                                __bf16* __restrict__ Ybf, int Bn, int C, int Din, int Dout) {
  const int Dout2 = Dout * Dout, Dout3 = Dout2 * Dout;
  const int total = Bn * Dout3 * C;
  const int idx = blockIdx.x * blockDim.x + threadIdx.x;
  if (idx >= total) return;
  const int c = idx % C;
  int n = idx / C;
  int b = n / Dout3;
  int r = n - b * Dout3;
  int od = r / Dout2; r -= od * Dout2;
  int oh = r / Dout;
  int ow = r - oh * Dout;
  const float* Xp = X + ((size_t)b * Din * Din * Din) * C + c;
  float m = -3.4e38f;
  for (int kd = 0; kd < 3; ++kd) {
    int id = od * 2 - 1 + kd;
    if ((unsigned)id >= (unsigned)Din) continue;
    for (int kh = 0; kh < 3; ++kh) {
      int ih = oh * 2 - 1 + kh;
      if ((unsigned)ih >= (unsigned)Din) continue;
      for (int kw = 0; kw < 3; ++kw) {
        int iw = ow * 2 - 1 + kw;
        if ((unsigned)iw >= (unsigned)Din) continue;
        m = fmaxf(m, Xp[(size_t)((id * Din + ih) * Din + iw) * C]);
      }
    }
  }
  Yf[idx] = m;
  if (Ybf) Ybf[idx] = f2bf(m);
}

// Global average pool (NDHWC): [B][S][C] -> [B][C]
__global__ void avgpool_kernel(const float* __restrict__ X, float* __restrict__ out,
                               int Bn, int S, int C) {
  int i = blockIdx.x * blockDim.x + threadIdx.x;
  if (i >= Bn * C) return;
  int b = i / C, c = i % C;
  float s = 0.f;
  for (int k = 0; k < S; ++k) s += X[((size_t)b * S + k) * C + c];
  out[i] = s / (float)S;
}

__global__ void fc_kernel(const float* __restrict__ pooled, const float* __restrict__ w,
                          const float* __restrict__ bias, float* __restrict__ out) {
  int i = threadIdx.x;
  if (i >= 16) return;
  int b = i >> 2, o = i & 3;
  float s = bias[o];
  for (int k = 0; k < 512; ++k) s += pooled[b * 512 + k] * w[o * 512 + k];
  out[i] = s;
}

// ---------------------------------------------------------------------------
// Host orchestration.  Input flattening in source-dict insertion order:
//  0:x 1:conv1 2:bn1_g 3:bn1_b
//  per block L: conv1, bn1_g, bn1_b, conv2, bn2_g, bn2_b [, down_w, down_g, down_b]
//  37:fc_w 38:fc_b
// ---------------------------------------------------------------------------
extern "C" void kernel_launch(void* const* d_in, const int* in_sizes, int n_in,
                              void* d_out, int out_size, void* d_ws, size_t ws_size,
                              hipStream_t stream) {
  (void)in_sizes; (void)n_in; (void)out_size; (void)ws_size;
  char* ws = (char*)d_ws;
  size_t off = 0;
  auto salloc = [&](size_t bytes) -> char* {
    off = (off + 255) & ~(size_t)255;
    char* p = ws + off;
    off += bytes;
    return p;
  };

  float*  colsum   = (float*)salloc((size_t)NATOM * 4);
  __bf16* fieldsBF = (__bf16*)salloc((size_t)4 * GRID3 * 32 * 2);   // NDHWC, CinP=32
  float*  y1       = (float*)salloc((size_t)4 * 27000 * 64 * 4);    // conv1 out NDHWC
  __bf16* act1bf   = (__bf16*)salloc((size_t)4 * 27000 * 64 * 2);
  float*  poolF    = (float*)salloc((size_t)4 * 3375 * 64 * 4);
  __bf16* poolBF   = (__bf16*)salloc((size_t)4 * 3375 * 64 * 2);
  float*  bufA     = (float*)salloc((size_t)864000 * 4);
  float*  bufB     = (float*)salloc((size_t)864000 * 4);
  float*  bufD     = (float*)salloc((size_t)262144 * 4);
  __bf16* bfA      = (__bf16*)salloc((size_t)864000 * 2);
  float*  stats    = (float*)salloc((size_t)512 * 2 * 4);
  float*  actF0    = (float*)salloc((size_t)864000 * 4);
  float*  actF1    = (float*)salloc((size_t)864000 * 4);
  __bf16* actBF0   = (__bf16*)salloc((size_t)864000 * 2);
  __bf16* actBF1   = (__bf16*)salloc((size_t)864000 * 2);
  float*  pooled   = (float*)salloc((size_t)4 * 512 * 4);
  __bf16* wbf      = (__bf16*)salloc((size_t)15100000 * 2);
  float*  actF[2]  = {actF0, actF1};
  __bf16* actBF[2] = {actBF0, actBF1};

  size_t wpos = 0;
  auto stage_w = [&](const void* src, int Cout, int Cin, int CinP, int k3) -> __bf16* {
    __bf16* dst = wbf + wpos;
    int n = Cout * k3 * CinP;
    wpos += (size_t)n;
    stage_weights_kernel<<<dim3((n + 255) / 256), dim3(256), 0, stream>>>(
        (const float*)src, dst, Cout, Cin, CinP, k3);
    return dst;
  };
  auto launch_conv = [&](const __bf16* X, const __bf16* W, float* Y, int Bn, int CinP,
                         int Din, int Cout, int Dout, int k, int s, int p) {
    int mtiles2 = Cout / 32;
    int ntiles2 = (Bn * Dout * Dout * Dout + 31) / 32;
    int tiles   = mtiles2 * ntiles2;
    int blocks  = (tiles + 3) / 4;   // 4 waves per 128-thread block
    conv3d_wmma_kernel<<<dim3(blocks), dim3(128), 0, stream>>>(X, W, Y, Bn, CinP, Din,
                                                               Cout, Dout, k, s, p);
  };

  const float* xin = (const float*)d_in[0];

  // ---- Gaussian blurring -> NDHWC bf16 fields [4][60^3][32] ----
  colsum_kernel<<<dim3(NATOM), dim3(256), 0, stream>>>(xin, colsum);
  fields_kernel<<<dim3((GRID3 + 127) / 128), dim3(128), 0, stream>>>(xin, colsum, fieldsBF);

  // ---- Stem: conv1 7^3 s2 p3 (4->64, 60^3->30^3), BN, leaky(0.1), maxpool ----
  __bf16* w1 = stage_w(d_in[1], 64, 4, 32, 343);
  launch_conv(fieldsBF, w1, y1, 4, 32, 60, 64, 30, 7, 2, 3);
  bn_stats_kernel<<<dim3(64), dim3(256), 0, stream>>>(y1, 4, 64, 27000, stats);
  {
    int tot = 4 * 27000 * 64;
    bn_act_kernel<<<dim3((tot + 255) / 256), dim3(256), 0, stream>>>(
        y1, (const float*)d_in[2], (const float*)d_in[3], stats, (const float*)nullptr,
        y1, act1bf, 64, tot, 2);
  }
  {
    int tot = 4 * 3375 * 64;
    maxpool3_kernel<<<dim3((tot + 255) / 256), dim3(256), 0, stream>>>(
        y1, poolF, poolBF, 4, 64, 30, 15);
  }

  // ---- Residual blocks ----
  const int idx[4][9] = {
      {4, 5, 6, 7, 8, 9, -1, -1, -1},
      {10, 11, 12, 13, 14, 15, 16, 17, 18},
      {19, 20, 21, 22, 23, 24, 25, 26, 27},
      {28, 29, 30, 31, 32, 33, 34, 35, 36}};
  const int outCs[4]   = {64, 128, 256, 512};
  const int strides[4] = {1, 2, 2, 2};

  float*  cur_f  = poolF;
  __bf16* cur_bf = poolBF;
  int curC = 64, curD = 15;

  for (int L = 0; L < 4; ++L) {
    const int Cout = outCs[L];
    const int s    = strides[L];
    const int Dout = (s == 1) ? curD : ((curD - 1) / 2 + 1);
    const int S    = Dout * Dout * Dout;
    const int tot  = 4 * S * Cout;
    const int gb   = (tot + 255) / 256;

    // conv1 + BN + relu
    __bf16* wc1 = stage_w(d_in[idx[L][0]], Cout, curC, curC, 27);
    launch_conv(cur_bf, wc1, bufA, 4, curC, curD, Cout, Dout, 3, s, 1);
    bn_stats_kernel<<<dim3(Cout), dim3(256), 0, stream>>>(bufA, 4, Cout, S, stats);
    bn_act_kernel<<<dim3(gb), dim3(256), 0, stream>>>(
        bufA, (const float*)d_in[idx[L][1]], (const float*)d_in[idx[L][2]], stats,
        (const float*)nullptr, bufA, bfA, Cout, tot, 1);

    // conv2 + BN (no act yet)
    __bf16* wc2 = stage_w(d_in[idx[L][3]], Cout, Cout, Cout, 27);
    launch_conv(bfA, wc2, bufB, 4, Cout, Dout, Cout, Dout, 3, 1, 1);

    // identity path
    const float* resid = cur_f;
    if (idx[L][6] >= 0) {
      __bf16* wd = stage_w(d_in[idx[L][6]], Cout, curC, curC, 1);
      launch_conv(cur_bf, wd, bufD, 4, curC, curD, Cout, Dout, 1, s, 0);
      bn_stats_kernel<<<dim3(Cout), dim3(256), 0, stream>>>(bufD, 4, Cout, S, stats);
      bn_act_kernel<<<dim3(gb), dim3(256), 0, stream>>>(
          bufD, (const float*)d_in[idx[L][7]], (const float*)d_in[idx[L][8]], stats,
          (const float*)nullptr, bufD, (__bf16*)nullptr, Cout, tot, 0);
      resid = bufD;
    }

    // out = relu(bn(conv2) + identity)
    bn_stats_kernel<<<dim3(Cout), dim3(256), 0, stream>>>(bufB, 4, Cout, S, stats);
    float*  of  = actF[L & 1];
    __bf16* obf = actBF[L & 1];
    bn_act_kernel<<<dim3(gb), dim3(256), 0, stream>>>(
        bufB, (const float*)d_in[idx[L][4]], (const float*)d_in[idx[L][5]], stats,
        resid, of, obf, Cout, tot, 1);

    cur_f  = of;
    cur_bf = obf;
    curC   = Cout;
    curD   = Dout;
  }

  // ---- Global average pool + FC ----
  avgpool_kernel<<<dim3((4 * 512 + 255) / 256), dim3(256), 0, stream>>>(
      cur_f, pooled, 4, curD * curD * curD, 512);
  fc_kernel<<<dim3(1), dim3(16), 0, stream>>>(pooled, (const float*)d_in[37],
                                              (const float*)d_in[38], (float*)d_out);
}
